// Embeddings_62440234549511
// MI455X (gfx1250) — matrix operand order repaired and checked by reading, not compiled, not run on hardware
//
#include <hip/hip_runtime.h>

#define N_VEC    8192
#define DIM      512
#define BATCH_SZ 64
#define COS_THR  0.8f
#define NWORDS16 (N_VEC / 16)   // 512 uint16 mask words per row
#define TILE_K   32
#define BLK      128
#define LDS_STRIDE 80           // 32 bf16 = 64B data + 16B pad (16B aligned, conflict-free)
#define MAXM     64             // per-row match-list capacity

typedef __bf16 bf16;
typedef __attribute__((ext_vector_type(16))) __bf16 v16bf;
typedef __attribute__((ext_vector_type(8)))  __bf16 v8bf;
typedef __attribute__((ext_vector_type(8)))  float  v8f;

// ---- CDNA5 async global->LDS copy (ASYNCcnt path), with safe fallback ------
#if defined(__has_builtin)
# if __has_builtin(__builtin_amdgcn_global_load_async_to_lds_b128) && \
     __has_builtin(__builtin_amdgcn_s_wait_asynccnt)
#  define HAVE_ASYNC 1
# endif
#endif
#ifndef HAVE_ASYNC
# define HAVE_ASYNC 0
#endif

// exact pointee type from the hipcc diagnostic: int vector_size(16)
typedef int v4i_ __attribute__((vector_size(4 * sizeof(int))));
typedef __attribute__((address_space(1))) v4i_* gp4_t;   // prints as "v4i __device__ *"
typedef __attribute__((address_space(3))) v4i_* lp4_t;   // LDS

__device__ __forceinline__ void copy16_g2l(void* l, const void* g)
{
#if HAVE_ASYNC
    // LDS offset = low 32 bits of the flat address; global keeps full 64-bit VA
    __builtin_amdgcn_global_load_async_to_lds_b128(
        (gp4_t)(unsigned long long)(size_t)g,
        (lp4_t)(unsigned int)(size_t)l,
        0, 0);
#else
    *(int4*)l = *(const int4*)g;   // global_load_b128 + ds_store_b128
#endif
}

// ---------------------------------------------------------------------------
// Kernel 1: f32 -> bf16 (round-to-nearest-even)
// ---------------------------------------------------------------------------
__global__ void cvt_bf16_kernel(const float* __restrict__ V,
                                unsigned short* __restrict__ Vh, int n)
{
    int i = blockIdx.x * blockDim.x + threadIdx.x;
    if (i < n) {
        unsigned int u = __float_as_uint(V[i]);
        Vh[i] = (unsigned short)((u + 0x7FFFu + ((u >> 16) & 1u)) >> 16);
    }
}

// ---------------------------------------------------------------------------
// Kernel 2: thresholded-similarity bitmask, LDS-tiled double-buffered WMMA.
// Block: 256 threads = 8 waves -> 128x128 output tile; wave: 32(M) x 64(N).
// ---------------------------------------------------------------------------
__global__ void __launch_bounds__(256)
sim_mask_kernel(const bf16* __restrict__ Vh, unsigned short* __restrict__ Mask)
{
    const int lane   = threadIdx.x & 31;
    const int wave   = threadIdx.x >> 5;   // 0..7
    const int waveM  = wave >> 1;          // 0..3  (32-row slab)
    const int waveN  = wave & 1;           // 0..1  (64-col slab)
    const int half16 = lane & 15;

    const int bi = blockIdx.x >> 6;
    const int bj = blockIdx.x & 63;
    const int i0 = bi << 7;
    const int j0 = bj << 7;

    if (j0 < i0) {
        // whole 128x128 block fails jmask (all j <= bs): zero-fill mask words
        const int t = threadIdx.x, row = t >> 1, hw = t & 1;
        *(unsigned long long*)(Mask + (size_t)(i0 + row) * NWORDS16 + (j0 >> 4) + hw * 4) = 0ull;
        return;
    }

    __shared__ unsigned char sA[2][BLK * LDS_STRIDE];   // 2 x 10 KB
    __shared__ unsigned char sB[2][BLK * LDS_STRIDE];   // 2 x 10 KB

    // Fill one K-slice (128 rows x 32 bf16) of A and B into LDS buffer `buf`.
    // 4 async b128 instructions per wave per call (2 for A, 2 for B).
    auto fill = [&](int buf, int k0) {
        const int t = threadIdx.x;
#pragma unroll
        for (int it = 0; it < 2; ++it) {
            int c = t + it * 256;            // 0..511 chunk id
            int row = c >> 2, q = c & 3;     // 16B chunk q of row
            copy16_g2l(&sA[buf][row * LDS_STRIDE + q * 16],
                       Vh + (size_t)(i0 + row) * DIM + k0 + q * 8);
        }
#pragma unroll
        for (int it = 0; it < 2; ++it) {
            int c = t + it * 256;
            int row = c >> 2, q = c & 3;
            copy16_g2l(&sB[buf][row * LDS_STRIDE + q * 16],
                       Vh + (size_t)(j0 + row) * DIM + k0 + q * 8);
        }
    };

    v8f acc[2][4] = {};

    fill(0, 0);
    fill(1, TILE_K);

    // Fragment addressing (contiguous k order, same for A and B):
    // A 16x32: lane<16 -> K k0..k0+15 ; lane>=16 -> K k0+16..k0+31
    // B 32x16: lane<16 -> K k0..k0+15 ; lane>=16 -> K k0+16..k0+31
    const int rA  = waveM * 32 + half16;
    const int kbA = (lane < 16) ? 0 : 16;
    const int kbB = (lane < 16) ? 0 : 16;

    for (int t = 0; t < DIM / TILE_K; ++t) {
        const int buf = t & 1;
#if HAVE_ASYNC
        if (t + 2 < DIM / TILE_K) __builtin_amdgcn_s_wait_asynccnt(4);
        else                      __builtin_amdgcn_s_wait_asynccnt(0);
#endif
        __syncthreads();   // buffer `buf` fully resident for all waves

        const unsigned char* a0p = &sA[buf][(rA)      * LDS_STRIDE + kbA * 2];
        const unsigned char* a1p = &sA[buf][(rA + 16) * LDS_STRIDE + kbA * 2];
        v16bf a0, a1;
        {
            v8bf lo0 = *(const v8bf*)a0p, hi0 = *(const v8bf*)(a0p + 16);
            v8bf lo1 = *(const v8bf*)a1p, hi1 = *(const v8bf*)(a1p + 16);
#pragma unroll
            for (int e = 0; e < 8; ++e) {
                a0[e] = lo0[e]; a0[e + 8] = hi0[e];
                a1[e] = lo1[e]; a1[e + 8] = hi1[e];
            }
        }
#pragma unroll
        for (int s = 0; s < 4; ++s) {
            const int col = waveN * 64 + s * 16 + half16;
            v16bf b = *(const v16bf*)&sB[buf][col * LDS_STRIDE + kbB * 2];
            acc[0][s] = __builtin_amdgcn_wmma_f32_16x16x32_bf16(
                false, a0, false, b, (short)0, acc[0][s], false, false);
            acc[1][s] = __builtin_amdgcn_wmma_f32_16x16x32_bf16(
                false, a1, false, b, (short)0, acc[1][s], false, false);
        }
        __syncthreads();   // all waves done reading `buf`
        if (t + 2 < DIM / TILE_K) fill(buf, (t + 2) * TILE_K);
    }

    // Threshold + jmask -> 16-bit masks via wave ballots.
    // C layout: VGPR r = row r (lanes 0-15, col=lane) / row r+8 (lanes 16-31).
#pragma unroll
    for (int m = 0; m < 2; ++m) {
        const int rbase = i0 + waveM * 32 + m * 16;
        const int bs    = rbase & ~63;
#pragma unroll
        for (int s = 0; s < 4; ++s) {
            const int jt  = j0 + waveN * 64 + s * 16;
            const bool jok = (jt + half16) > bs;
            unsigned int my = 0;
#pragma unroll
            for (int r = 0; r < 8; ++r) {
                unsigned int bal =
                    __builtin_amdgcn_ballot_w32(jok && (acc[m][s][r] >= COS_THR));
                if ((lane & 7) == r) my = bal;
            }
            if (lane < 16) {
                unsigned short v = (lane < 8) ? (unsigned short)(my & 0xFFFFu)
                                              : (unsigned short)(my >> 16);
                Mask[(size_t)(rbase + lane) * NWORDS16 + (jt >> 4)] = v;
            }
        }
    }
}

// ---------------------------------------------------------------------------
// Kernel 3: compact the sparse mask into per-row match lists (parallel).
// ---------------------------------------------------------------------------
__global__ void __launch_bounds__(128)
build_lists_kernel(const unsigned short* __restrict__ Mask,
                   unsigned short* __restrict__ mlist,
                   int* __restrict__ mcnt)
{
    const int row = blockIdx.x;
    const int t   = threadIdx.x;
    __shared__ int c;
    if (t == 0) c = 0;
    __syncthreads();
    for (int w = t; w < NWORDS16; w += 128) {
        unsigned int m = Mask[(size_t)row * NWORDS16 + w];
        while (m) {
            int bit = __ffs(m) - 1;
            m &= m - 1;
            int pos = atomicAdd(&c, 1);
            if (pos < MAXM) mlist[row * MAXM + pos] = (unsigned short)((w << 4) + bit);
        }
    }
    __syncthreads();
    if (t == 0) mcnt[row] = c;   // may exceed MAXM -> merge falls back to scan
}

// ---------------------------------------------------------------------------
// Kernel 4: sequential transitive merge, faithful to the reference scan.
// Rows with no raw matches skip all barriers (uniform branch).
// ---------------------------------------------------------------------------
__global__ void __launch_bounds__(512)
merge_kernel(const unsigned short* __restrict__ Mask,
             const unsigned short* __restrict__ mlist,
             const int* __restrict__ mcnt,
             const int* __restrict__ g_in,
             int* __restrict__ g_out)
{
    extern __shared__ int smem[];
    int* g  = smem;                                         // [N_VEC]
    int* ge = smem + N_VEC;                                 // [N_VEC] batch-entry snapshot
    unsigned int* flag = (unsigned int*)(smem + 2 * N_VEC); // [N_VEC/32] group bitmap
    int* anyf = (int*)(flag + N_VEC / 32);                  // [2] active-row parity slots

    const int tid = threadIdx.x;
    const int NT  = 512;

    for (int i = tid; i < N_VEC; i += NT) g[i] = g_in[i];
    for (int w = tid; w < N_VEC / 32; w += NT) flag[w] = 0;
    if (tid < 2) anyf[tid] = 0;
    __syncthreads();

    int arow = 0;   // active (non-empty) row counter, identical on all threads

    for (int b = 0; b < N_VEC / BATCH_SZ; ++b) {
        const int bs = b * BATCH_SZ;
        for (int i = tid; i < N_VEC; i += NT) ge[i] = g[i];
        __syncthreads();

        for (int k = 0; k < BATCH_SZ; ++k) {
            const int row = bs + k;
            const int cnt = mcnt[row];        // uniform load -> uniform branch
            if (cnt == 0) continue;           // no LDS access, no barrier

            const int par = (arow++) & 1;
            const int gbk = ge[row];          // batch-entry group of source row
            const int tgt = g[row];           // CURRENT group (scan-carried)

            bool did = false;
            if (cnt <= MAXM) {
                if (tid < cnt) {
                    int j = mlist[row * MAXM + tid];
                    if (ge[j] != gbk) {
                        int gj = g[j];
                        atomicOr(&flag[gj >> 5], 1u << (gj & 31));
                        did = true;
                    }
                }
            } else {  // overflow fallback: full mask-row scan (tid < 512 == NWORDS16)
                unsigned int m = Mask[(size_t)row * NWORDS16 + tid];
                while (m) {
                    int bit = __ffs(m) - 1;
                    m &= m - 1;
                    int j = (tid << 4) + bit;
                    if (ge[j] != gbk) {
                        int gj = g[j];
                        atomicOr(&flag[gj >> 5], 1u << (gj & 31));
                        did = true;
                    }
                }
            }
            if (did) anyf[par] = 1;
            __syncthreads();

            if (anyf[par]) {
                for (int i = tid; i < N_VEC; i += NT) {
                    int gi = g[i];
                    if ((flag[gi >> 5] >> (gi & 31)) & 1u) g[i] = tgt;
                }
                __syncthreads();
                if (tid == 0) anyf[par] = 0;  // all readers passed the barrier
                for (int w = tid; w < N_VEC / 32; w += NT) flag[w] = 0;
                __syncthreads();
            }
        }
    }
    __syncthreads();
    for (int i = tid; i < N_VEC; i += NT) g_out[i] = g[i];
}

// ---------------------------------------------------------------------------
extern "C" void kernel_launch(void* const* d_in, const int* in_sizes, int n_in,
                              void* d_out, int out_size, void* d_ws, size_t ws_size,
                              hipStream_t stream)
{
    const float* V  = (const float*)d_in[0];   // [N_VEC*DIM] f32
    const int* g_in = (const int*)d_in[1];     // [N_VEC] initial group ids
    int* g_out      = (int*)d_out;             // [N_VEC] final group ids

    // workspace layout
    char* ws = (char*)d_ws;
    unsigned short* Vh    = (unsigned short*)(ws);                                  //  8 MB
    unsigned short* Mask  = (unsigned short*)(ws + (size_t)N_VEC * DIM * 2);        //  8 MB
    unsigned short* mlist = (unsigned short*)(ws + (size_t)16 * 1024 * 1024);       //  1 MB
    int*            mcnt  = (int*)(ws + (size_t)17 * 1024 * 1024);                  // 32 KB

    const int nelem = N_VEC * DIM;
    cvt_bf16_kernel<<<(nelem + 255) / 256, 256, 0, stream>>>(V, Vh, nelem);

    // 64x64 blocks of 128x128 tiles
    sim_mask_kernel<<<64 * 64, 256, 0, stream>>>((const bf16*)Vh, Mask);

    build_lists_kernel<<<N_VEC, 128, 0, stream>>>(Mask, mlist, mcnt);

    const size_t smem = (size_t)(2 * N_VEC) * sizeof(int)
                      + (N_VEC / 32) * sizeof(unsigned int)
                      + 2 * sizeof(int);
    merge_kernel<<<1, 512, smem, stream>>>(Mask, mlist, mcnt, g_in, g_out);
}
